// Siamese_Model_Simple_RNN_25022479466788
// MI455X (gfx1250) — matrix operand order repaired and checked by reading, not compiled, not run on hardware
//
#include <hip/hip_runtime.h>
#include <hip/hip_bf16.h>
#include <math.h>

// ---------------------------------------------------------------------------
// Siamese SimpleRNN (mask_zero) + cosine similarity, CDNA5 / gfx1250 (wave32)
//   state = tanh(x@W + h@U + b), h kept when id==0
// GEMMs via v_wmma_f32_16x16x32_bf16; W/U pre-packed bf16 B-fragments
// (768 KB, L2-resident). f32 hidden state carried in C-fragments in VGPRs;
// bf16 copy cycles through LDS as next step's A-fragments.
// Embedding gather for step t+1 runs on the ASYNCcnt path
// (global_load_async_to_lds_b128) during step t's WMMA phase, so gather
// latency never interacts with the in-order LOADcnt waits of the B-fragment
// stream. kt-loops kept as real loops (unroll 2) to prevent LICM fragment
// hoisting/spilling (seen round 1).
// ---------------------------------------------------------------------------

typedef __attribute__((ext_vector_type(16))) __bf16 v16bf;
typedef __attribute__((ext_vector_type(8)))  __bf16 v8bf;
typedef __attribute__((ext_vector_type(8)))  float   v8f;

#define VOCABN 50001
#define EMBED  256
#define SEQ    64
#define FEAT   512
#define BATCH  1024
#define NT_F   (FEAT / 16)     // 32 N-tiles across FEAT
#define KT_E   (EMBED / 32)    // 8  K-tiles across EMBED
#define KT_F   (FEAT / 32)     // 16 K-tiles across FEAT
#define KT_STRIDE (NT_F * 32 * 16)   // bf16 elems per kt block in packed B

// ---------------------------------------------------------------------------
// Pack f32 [K][N] row-major matrix into bf16 WMMA B-fragment order:
//   dst[((kt*(N/16) + nt)*32 + lane)*16 + e]  where
//   k = kt*32 + (e>=8 ? 16 : 0) + (lane>=16 ? 8 : 0) + (e&7),  n = nt*16 + (lane&15)
// ---------------------------------------------------------------------------
__global__ void pack_b_kernel(const float* __restrict__ src,
                              __bf16* __restrict__ dst, int K, int N) {
  int idx = blockIdx.x * blockDim.x + threadIdx.x;
  int ntiles = N / 16;
  int total = (K / 32) * ntiles * 32 * 16;
  if (idx >= total) return;
  int e    = idx & 15;
  int lane = (idx >> 4) & 31;
  int knt  = idx >> 9;          // kt*ntiles + nt
  int nt   = knt % ntiles;
  int kt   = knt / ntiles;
  int k = kt * 32 + ((e >> 3) << 4) + ((lane >> 4) << 3) + (e & 7);
  int n = nt * 16 + (lane & 15);
  dst[idx] = (__bf16)src[(size_t)k * N + n];
}

// Async copy of 64 contiguous bytes (global -> LDS) on the ASYNCcnt path.
// INST_OFFSET is added to both the LDS and global addresses.
__device__ __forceinline__ void async_copy64(unsigned ldsDst, const float* gsrc) {
  asm volatile("global_load_async_to_lds_b128 %0, %1, off"
               :: "v"(ldsDst), "v"(gsrc) : "memory");
  asm volatile("global_load_async_to_lds_b128 %0, %1, off offset:16"
               :: "v"(ldsDst), "v"(gsrc) : "memory");
  asm volatile("global_load_async_to_lds_b128 %0, %1, off offset:32"
               :: "v"(ldsDst), "v"(gsrc) : "memory");
  asm volatile("global_load_async_to_lds_b128 %0, %1, off offset:48"
               :: "v"(ldsDst), "v"(gsrc) : "memory");
}

// ---------------------------------------------------------------------------
// Fused gather + SimpleRNN recurrence. Grid: (BATCH/16, 2 branches), 256 thr.
// ---------------------------------------------------------------------------
__global__ __launch_bounds__(256) void rnn_kernel(
    const int* __restrict__ ids1, const int* __restrict__ ids2,
    const float* __restrict__ emb,
    const __bf16* __restrict__ Wp,   // [KT_E][NT_F][32][16] bf16
    const __bf16* __restrict__ Up,   // [KT_F][NT_F][32][16] bf16
    const float* __restrict__ bias,  // [FEAT]
    float* __restrict__ st1, float* __restrict__ st2) {

  __shared__ __align__(16) int    ids_s[16 * SEQ];
  __shared__ __align__(16) float  xf_s[2][16 * EMBED];  // async f32 staging
  __shared__ __align__(16) __bf16 x_s[16 * EMBED];
  __shared__ __align__(16) __bf16 h_s[16 * FEAT];
  __shared__ __align__(16) float  bias_s[FEAT];

  const int tid    = threadIdx.x;
  const int lane   = tid & 31;
  const int wave   = tid >> 5;
  const int laneHi = lane >> 4;     // 0: lanes 0-15, 1: lanes 16-31
  const int nlo    = lane & 15;
  const int row0   = blockIdx.x * 16;
  const int branch = blockIdx.y;

  const int*  ids = branch ? ids2 : ids1;
  float* state_out = branch ? st2 : st1;

  // Per-lane base pointers into packed B operands (wave owns 4 N-tiles).
  const __bf16* WpL = Wp + (size_t)(wave * 4) * 512 + lane * 16;
  const __bf16* UpL = Up + (size_t)(wave * 4) * 512 + lane * 16;
  // Per-lane A-fragment row bases in LDS.
  const __bf16* xrow = x_s + nlo * EMBED + laneHi * 8;
  const __bf16* hrow = h_s + nlo * FEAT  + laneHi * 8;

  // This thread's 16-element gather slab.
  const int m  = tid >> 4;             // row within 16-row tile
  const int c0 = (tid & 15) << 4;      // 16-column slab

  // Prologue: stage token ids [16][SEQ], bias, zero h.
  for (int j = tid; j < 16 * SEQ; j += 256)
    ids_s[j] = ids[(size_t)(row0 + (j >> 6)) * SEQ + (j & 63)];
  for (int j = tid; j < FEAT; j += 256)
    bias_s[j] = bias[j];
  for (int j = tid; j < 16 * FEAT; j += 256)
    h_s[j] = (__bf16)0.0f;

  // Persistent f32 hidden-state fragments: 4 N-tiles per wave (C-layout).
  v8f hprev[4];
#pragma unroll
  for (int i = 0; i < 4; ++i)
#pragma unroll
    for (int r = 0; r < 8; ++r) hprev[i][r] = 0.0f;

  __syncthreads();

  // Kick off async gather for t=0 into xf_s[0].
  {
    int id0 = ids_s[m * SEQ + 0];
    async_copy64((unsigned)(uintptr_t)&xf_s[0][m * EMBED + c0],
                 emb + (size_t)id0 * EMBED + c0);
  }

  for (int t = 0; t < SEQ; ++t) {
    const int cur = t & 1;

    // ---- wait for this step's async gather; convert f32 -> bf16 in x_s ----
    asm volatile("s_wait_asynccnt 0" ::: "memory");
    {
      const float4* s4 = (const float4*)&xf_s[cur][m * EMBED + c0];
      float4 f0 = s4[0], f1 = s4[1], f2 = s4[2], f3 = s4[3];
      v8bf o0, o1;
      o0[0] = (__bf16)f0.x; o0[1] = (__bf16)f0.y;
      o0[2] = (__bf16)f0.z; o0[3] = (__bf16)f0.w;
      o0[4] = (__bf16)f1.x; o0[5] = (__bf16)f1.y;
      o0[6] = (__bf16)f1.z; o0[7] = (__bf16)f1.w;
      o1[0] = (__bf16)f2.x; o1[1] = (__bf16)f2.y;
      o1[2] = (__bf16)f2.z; o1[3] = (__bf16)f2.w;
      o1[4] = (__bf16)f3.x; o1[5] = (__bf16)f3.y;
      o1[6] = (__bf16)f3.z; o1[7] = (__bf16)f3.w;
      *(v8bf*)&x_s[m * EMBED + c0]     = o0;
      *(v8bf*)&x_s[m * EMBED + c0 + 8] = o1;
    }
    // ---- launch async gather for t+1 (stays in flight across WMMA phase) --
    {
      int tn  = (t + 1 < SEQ) ? t + 1 : t;     // clamped; tail refetch is harmless
      int idn = ids_s[m * SEQ + tn];
      async_copy64((unsigned)(uintptr_t)&xf_s[cur ^ 1][m * EMBED + c0],
                   emb + (size_t)idn * EMBED + c0);
    }
    __syncthreads();

    // ---- accumulators: init with bias ----
    v8f acc[4];
#pragma unroll
    for (int i = 0; i < 4; ++i) {
      float bb = bias_s[(wave * 4 + i) * 16 + nlo];
#pragma unroll
      for (int r = 0; r < 8; ++r) acc[i][r] = bb;
    }

    // ---- x @ W : 8 K-tiles over EMBED (real loop: no LICM spills) ----
#pragma unroll 2
    for (int kt = 0; kt < KT_E; ++kt) {
      v8bf lo = *(const v8bf*)(xrow + kt * 32);
      v8bf hi = *(const v8bf*)(xrow + kt * 32 + 16);
      v16bf a = __builtin_shufflevector(lo, hi, 0, 1, 2, 3, 4, 5, 6, 7,
                                        8, 9, 10, 11, 12, 13, 14, 15);
      const __bf16* bp = WpL + (size_t)kt * KT_STRIDE;
#pragma unroll
      for (int i = 0; i < 4; ++i) {
        v16bf bf = *(const v16bf*)(bp + i * 512);
        acc[i] = __builtin_amdgcn_wmma_f32_16x16x32_bf16(
            false, a, false, bf, (short)0, acc[i], false, false);
      }
    }

    // ---- h @ U : 16 K-tiles over FEAT ----
#pragma unroll 2
    for (int kt = 0; kt < KT_F; ++kt) {
      v8bf lo = *(const v8bf*)(hrow + kt * 32);
      v8bf hi = *(const v8bf*)(hrow + kt * 32 + 16);
      v16bf a = __builtin_shufflevector(lo, hi, 0, 1, 2, 3, 4, 5, 6, 7,
                                        8, 9, 10, 11, 12, 13, 14, 15);
      const __bf16* bp = UpL + (size_t)kt * KT_STRIDE;
#pragma unroll
      for (int i = 0; i < 4; ++i) {
        v16bf bf = *(const v16bf*)(bp + i * 512);
        acc[i] = __builtin_amdgcn_wmma_f32_16x16x32_bf16(
            false, a, false, bf, (short)0, acc[i], false, false);
      }
    }
    __syncthreads();   // all reads of x_s/h_s for step t complete

    // ---- tanh + mask_zero select (exact f32 carry of kept state) ----
#pragma unroll
    for (int r = 0; r < 8; ++r) {
      int mm = r + laneHi * 8;       // C-layout: VGPR r -> row m (+8 hi lanes)
      bool upd = ids_s[mm * SEQ + t] != 0;
#pragma unroll
      for (int i = 0; i < 4; ++i) {
        float hv = tanhf(acc[i][r]);
        hprev[i][r] = upd ? hv : hprev[i][r];
      }
    }
    // publish bf16 copy as next step's A-operand
#pragma unroll
    for (int i = 0; i < 4; ++i) {
      int n = (wave * 4 + i) * 16 + nlo;
#pragma unroll
      for (int r = 0; r < 8; ++r)
        h_s[(r + laneHi * 8) * FEAT + n] = (__bf16)hprev[i][r];
    }
    __syncthreads();
  }

  // Epilogue: final state (f32) to output.
#pragma unroll
  for (int i = 0; i < 4; ++i) {
    int n = (wave * 4 + i) * 16 + nlo;
#pragma unroll
    for (int r = 0; r < 8; ++r)
      state_out[(size_t)(row0 + r + laneHi * 8) * FEAT + n] = hprev[i][r];
  }
}

// ---------------------------------------------------------------------------
// Cosine similarity per row: sim[b] = <n1, n2> with keras eps clamp.
// ---------------------------------------------------------------------------
__global__ __launch_bounds__(128) void cos_kernel(
    const float* __restrict__ s1, const float* __restrict__ s2,
    float* __restrict__ sim) {
  int b = blockIdx.x;
  int tid = threadIdx.x;
  float dp = 0.f, q1 = 0.f, q2 = 0.f;
  for (int f = tid; f < FEAT; f += 128) {
    float a = s1[(size_t)b * FEAT + f];
    float c = s2[(size_t)b * FEAT + f];
    dp += a * c; q1 += a * a; q2 += c * c;
  }
  __shared__ float r0[128], r1[128], r2[128];
  r0[tid] = dp; r1[tid] = q1; r2[tid] = q2;
  __syncthreads();
  for (int s = 64; s > 0; s >>= 1) {
    if (tid < s) { r0[tid] += r0[tid + s]; r1[tid] += r1[tid + s]; r2[tid] += r2[tid + s]; }
    __syncthreads();
  }
  if (tid == 0) {
    float n1 = sqrtf(fmaxf(r1[0], 1e-12f));
    float n2 = sqrtf(fmaxf(r2[0], 1e-12f));
    sim[b] = r0[0] / (n1 * n2);
  }
}

// ---------------------------------------------------------------------------
extern "C" void kernel_launch(void* const* d_in, const int* in_sizes, int n_in,
                              void* d_out, int out_size, void* d_ws, size_t ws_size,
                              hipStream_t stream) {
  (void)in_sizes; (void)n_in; (void)out_size; (void)ws_size;
  const int*   ids1 = (const int*)d_in[0];
  const int*   ids2 = (const int*)d_in[1];
  const float* emb  = (const float*)d_in[2];
  const float* W    = (const float*)d_in[3];
  const float* U    = (const float*)d_in[4];
  const float* bias = (const float*)d_in[5];

  float* out    = (float*)d_out;
  float* state1 = out;
  float* state2 = out + (size_t)BATCH * FEAT;
  float* sim    = out + (size_t)2 * BATCH * FEAT;

  // Workspace: packed bf16 W (256 KB) + packed bf16 U (512 KB)
  __bf16* Wp = (__bf16*)d_ws;
  const int wElems = KT_E * NT_F * 32 * 16;   // 131072
  const int uElems = KT_F * NT_F * 32 * 16;   // 262144
  __bf16* Up = Wp + wElems;

  pack_b_kernel<<<(wElems + 255) / 256, 256, 0, stream>>>(W, Wp, EMBED, FEAT);
  pack_b_kernel<<<(uElems + 255) / 256, 256, 0, stream>>>(U, Up, FEAT, FEAT);

  dim3 grid(BATCH / 16, 2);
  rnn_kernel<<<grid, 256, 0, stream>>>(ids1, ids2, emb, Wp, Up, bias,
                                       state1, state2);
  cos_kernel<<<BATCH, 128, 0, stream>>>(state1, state2, sim);
}
